// EGNN_36163624632810
// MI455X (gfx1250) — matrix operand order repaired and checked by reading, not compiled, not run on hardware
//
#include <hip/hip_runtime.h>
#include <hip/hip_bf16.h>

#define NN   10000
#define NE   320000
#define HIDF 128

typedef __attribute__((ext_vector_type(16))) _Float16 v16h;
typedef __attribute__((ext_vector_type(8)))  _Float16 v8h;
typedef __attribute__((ext_vector_type(8)))  float    v8f;

#define DEVINL __device__ __forceinline__

#if defined(__HIP_DEVICE_COMPILE__) && __has_builtin(__builtin_amdgcn_global_load_async_to_lds_b128)
#define HAS_ASYNC_LDS 1
#endif

// The builtin expects pointers to 16B int-vectors (per hipcc diagnostic):
//   param0: v4i in global address space, param1: v4i in LDS address space.
typedef int v4i_t __attribute__((vector_size(16)));
typedef __attribute__((address_space(1))) v4i_t gv4i;
typedef __attribute__((address_space(3))) v4i_t lv4i;

// async 16B global -> LDS copy (ASYNCcnt path); falls back to load+ds_store
DEVINL void async_cp16(const _Float16* g, _Float16* l) {
#ifdef HAS_ASYNC_LDS
  __builtin_amdgcn_global_load_async_to_lds_b128((gv4i*)g, (lv4i*)l, 0, 0);
#else
  *(uint4*)(void*)l = *(const uint4*)(const void*)g;
#endif
}

DEVINL void wait_async() {
#ifdef HAS_ASYNC_LDS
#if __has_builtin(__builtin_amdgcn_s_wait_asynccnt)
  __builtin_amdgcn_s_wait_asynccnt(0);
#else
  asm volatile("s_wait_asynccnt 0x0" ::: "memory");
#endif
#endif
}

DEVINL float silu_f(float x) { return x / (1.0f + __expf(-x)); }

// A-fragment (16x32 f16) from LDS, row-major [16][stride].
// ISA layout: lane L holds row m=L&15; elements 0..7 = K (k0 + 8*(L>>4)) + 0..7,
// elements 8..15 = same + 16  -> two 16B LDS loads.
DEVINL v16h lda(const _Float16* base, int stride, int k0, int lane) {
  int m  = lane & 15;
  int kh = k0 + ((lane >> 4) << 3);
  const _Float16* r = base + m * stride + kh;
  union { v16h v; v8h h[2]; } u;
  u.h[0] = *(const v8h*)(r);
  u.h[1] = *(const v8h*)(r + 16);
  return u.v;
}

// B-fragment (32x16 f16) from pre-repacked weights: frag-major, 512 f16 per frag,
// lane-contiguous 16 halves (32B) per lane.
DEVINL v16h ldb(const _Float16* wf, int frag, int lane) {
  return *(const v16h*)(wf + (((size_t)frag * 32 + (size_t)lane) << 4));
}

DEVINL v8f wmma32(v16h a, v16h b, v8f c) {
  return __builtin_amdgcn_wmma_f32_16x16x32_f16(false, a, false, b, (short)0, c,
                                                false, false);
}

// ---------------- weight repack: W[K][N] f32 -> fragment-native f16 ----------
// out[((kt*NT+nt)*32 + lane)*16 + e] = W[kt*32 + 8*(lane>>4) + (e&7) + 16*(e>>3)][nt*16 + (lane&15)]
__global__ void repack_w(const float* __restrict__ W, _Float16* __restrict__ out,
                         int KT, int NT) {
  int total = KT * NT * 512;
  int i = blockIdx.x * blockDim.x + threadIdx.x;
  if (i >= total) return;
  int e = i & 15, lane = (i >> 4) & 31, t = i >> 9;
  int nt = t % NT, kt = t / NT;
  int k = kt * 32 + ((lane >> 4) << 3) + (e & 7) + ((e >> 3) << 4);
  int n = nt * 16 + (lane & 15);
  out[i] = (_Float16)W[(size_t)k * (NT * 16) + n];
}

__global__ void cvt_f16(const float* __restrict__ s, _Float16* __restrict__ d, int n) {
  int i = blockIdx.x * blockDim.x + threadIdx.x;
  if (i < n) d[i] = (_Float16)s[i];
}

__global__ void deg_kernel(const int* __restrict__ dst, float* __restrict__ deg, int E) {
  int i = blockIdx.x * blockDim.x + threadIdx.x;
  if (i < E) atomicAdd(&deg[dst[i]], 1.0f);
}

// ---------------- embed: h = silu(x @ Wemb + b) ------------------------------
__global__ void __launch_bounds__(128) embed_kernel(
    const float* __restrict__ x, const _Float16* __restrict__ wf,
    const _Float16* __restrict__ bias,
    float* __restrict__ h, _Float16* __restrict__ h16, int n_tiles) {
  __shared__ __align__(16) _Float16 sA[4][16][72];
  const int wave = threadIdx.x >> 5, lane = threadIdx.x & 31;
  const int nl = lane & 15, hi = lane >> 4;
  int tile = blockIdx.x * 4 + wave;
  const bool active = tile < n_tiles;
  if (!active) tile = n_tiles - 1;
  const int row0 = tile * 16;

  #pragma unroll 4
  for (int e = 0; e < 16; ++e) {
    float2 v = *(const float2*)(x + (size_t)(row0 + e) * 64 + lane * 2);
    union { _Float16 hh[2]; unsigned u; } p;
    p.hh[0] = (_Float16)v.x; p.hh[1] = (_Float16)v.y;
    *(unsigned*)&sA[wave][e][lane * 2] = p.u;
  }
  __syncthreads();

  const v8f vz = {0.f,0.f,0.f,0.f,0.f,0.f,0.f,0.f};
  v8f acc[8];
  #pragma unroll
  for (int nt = 0; nt < 8; ++nt) acc[nt] = vz;
  #pragma unroll
  for (int kt = 0; kt < 2; ++kt) {
    v16h a = lda(&sA[wave][0][0], 72, kt * 32, lane);
    #pragma unroll
    for (int nt = 0; nt < 8; ++nt)
      acc[nt] = wmma32(a, ldb(wf, kt * 8 + nt, lane), acc[nt]);
  }
  if (active) {
    #pragma unroll
    for (int nt = 0; nt < 8; ++nt) {
      int n = nt * 16 + nl;
      float bb = (float)bias[n];
      #pragma unroll
      for (int r = 0; r < 8; ++r) {
        int row = row0 + r + hi * 8;
        float xv = silu_f(acc[nt][r] + bb);
        h[(size_t)row * HIDF + n] = xv;
        h16[(size_t)row * HIDF + n] = (_Float16)xv;
      }
    }
  }
}

// ---------------- fused edge MLP + scatter -----------------------------------
struct __align__(16) EdgeSmem {
  union {
    _Float16 ain[16][272];                             // 8704 B (K=256 staging)
    struct { _Float16 t1[16][136]; _Float16 t2[16][136]; } s;  // aliases ain
  } u;
  float ev[16][3];
  float d2s[16];
  int   sidx[16];
  int   didx[16];
  float cw[16];
};

__global__ void __launch_bounds__(128) edge_kernel(
    const int* __restrict__ src, const int* __restrict__ dst,
    const float* __restrict__ pos, const _Float16* __restrict__ h16,
    float* __restrict__ agg, float* __restrict__ cd,
    const _Float16* __restrict__ ew1f, const _Float16* __restrict__ ew1last,
    const _Float16* __restrict__ eb1,
    const _Float16* __restrict__ ew2f, const _Float16* __restrict__ eb2,
    const _Float16* __restrict__ cw1f, const _Float16* __restrict__ cb1,
    const _Float16* __restrict__ cw2v) {
  __shared__ EdgeSmem sm[4];
  const int wave = threadIdx.x >> 5, lane = threadIdx.x & 31;
  const int nl = lane & 15, hi = lane >> 4;
  EdgeSmem& S = sm[wave];
  const int te = (blockIdx.x * 4 + wave) * 16;

  if (lane < 16) {
    int e = te + lane;
    int s = src[e], d = dst[e];
    S.sidx[lane] = s; S.didx[lane] = d;
    float dx = pos[d * 3 + 0] - pos[s * 3 + 0];
    float dy = pos[d * 3 + 1] - pos[s * 3 + 1];
    float dz = pos[d * 3 + 2] - pos[s * 3 + 2];
    S.ev[lane][0] = dx; S.ev[lane][1] = dy; S.ev[lane][2] = dz;
    S.d2s[lane] = dx * dx + dy * dy + dz * dz;
  }
  __syncthreads();

  {  // gather [h[src] | h[dst]] (f16) via async global->LDS copies.
     // lanes 0-15 -> src half, 16-31 -> dst half; 16B per lane per edge.
    const int col = nl * 8;
    #pragma unroll 4
    for (int e = 0; e < 16; ++e) {
      int node = hi ? S.didx[e] : S.sidx[e];
      async_cp16(h16 + (size_t)node * HIDF + col, &S.u.ain[e][hi * 128 + col]);
    }
    wait_async();
  }
  __syncthreads();

  const v8f vz = {0.f,0.f,0.f,0.f,0.f,0.f,0.f,0.f};
  v8f acc[8];

  // GEMM1: edge_in[:,0:256] @ e_w1[0:256,:]  (8 kt x 8 nt WMMAs)
  #pragma unroll
  for (int nt = 0; nt < 8; ++nt) acc[nt] = vz;
  #pragma unroll
  for (int kt = 0; kt < 8; ++kt) {
    v16h a = lda(&S.u.ain[0][0], 272, kt * 32, lane);
    if (kt < 7) __builtin_prefetch(ew1f + ((size_t)(kt + 1) * 8 * 32 * 16), 0, 1);
    #pragma unroll
    for (int nt = 0; nt < 8; ++nt)
      acc[nt] = wmma32(a, ldb(ew1f, kt * 8 + nt, lane), acc[nt]);
  }
  // epilogue: + d2 * e_w1[256,:] + b1, silu -> t1 (f16, aliases ain)
  float d2v[8];
  #pragma unroll
  for (int r = 0; r < 8; ++r) d2v[r] = S.d2s[r + hi * 8];
  #pragma unroll
  for (int nt = 0; nt < 8; ++nt) {
    int n = nt * 16 + nl;
    float wl = (float)ew1last[n];
    float bb = (float)eb1[n];
    #pragma unroll
    for (int r = 0; r < 8; ++r) {
      float xv = silu_f(acc[nt][r] + d2v[r] * wl + bb);
      S.u.s.t1[r + hi * 8][n] = (_Float16)xv;
    }
  }
  __syncthreads();

  // GEMM2: ef = silu(t1 @ e_w2 + b2); scatter ef into agg[dst]
  #pragma unroll
  for (int nt = 0; nt < 8; ++nt) acc[nt] = vz;
  #pragma unroll
  for (int kt = 0; kt < 4; ++kt) {
    v16h a = lda(&S.u.s.t1[0][0], 136, kt * 32, lane);
    #pragma unroll
    for (int nt = 0; nt < 8; ++nt)
      acc[nt] = wmma32(a, ldb(ew2f, kt * 8 + nt, lane), acc[nt]);
  }
  int d8[8];
  #pragma unroll
  for (int r = 0; r < 8; ++r) d8[r] = S.didx[r + hi * 8];
  #pragma unroll
  for (int nt = 0; nt < 8; ++nt) {
    int n = nt * 16 + nl;
    float bb = (float)eb2[n];
    #pragma unroll
    for (int r = 0; r < 8; ++r) {
      float xv = silu_f(acc[nt][r] + bb);
      S.u.s.t2[r + hi * 8][n] = (_Float16)xv;
      atomicAdd(&agg[(size_t)d8[r] * HIDF + n], xv);
    }
  }
  __syncthreads();

  // GEMM3: c1 = silu(ef @ c_w1 + cb1); cw = c1 @ c_w2  (dot via shuffle reduce)
  #pragma unroll
  for (int nt = 0; nt < 8; ++nt) acc[nt] = vz;
  #pragma unroll
  for (int kt = 0; kt < 4; ++kt) {
    v16h a = lda(&S.u.s.t2[0][0], 136, kt * 32, lane);
    #pragma unroll
    for (int nt = 0; nt < 8; ++nt)
      acc[nt] = wmma32(a, ldb(cw1f, kt * 8 + nt, lane), acc[nt]);
  }
  float p[8] = {0.f,0.f,0.f,0.f,0.f,0.f,0.f,0.f};
  #pragma unroll
  for (int nt = 0; nt < 8; ++nt) {
    int n = nt * 16 + nl;
    float bb = (float)cb1[n];
    float w2 = (float)cw2v[n];
    #pragma unroll
    for (int r = 0; r < 8; ++r)
      p[r] += silu_f(acc[nt][r] + bb) * w2;
  }
  #pragma unroll
  for (int r = 0; r < 8; ++r) {
    float v = p[r];
    v += __shfl_xor(v, 1);
    v += __shfl_xor(v, 2);
    v += __shfl_xor(v, 4);
    v += __shfl_xor(v, 8);
    if (nl == 0) S.cw[hi * 8 + r] = v;
  }
  __syncthreads();
  if (lane < 16) {
    float c = S.cw[lane];
    int d = S.didx[lane];
    atomicAdd(&cd[d * 3 + 0], S.ev[lane][0] * c);
    atomicAdd(&cd[d * 3 + 1], S.ev[lane][1] * c);
    atomicAdd(&cd[d * 3 + 2], S.ev[lane][2] * c);
  }
}

// ---------------- node update + residual + pos update ------------------------
struct __align__(16) NodeSmem {
  union {
    _Float16 ain[16][264];   // [h | agg] staging, K=256
    _Float16 t1[16][136];    // aliases
  } u;
};

__global__ void __launch_bounds__(128) node_kernel(
    float* __restrict__ h, _Float16* __restrict__ h16,
    const float* __restrict__ agg, const float* __restrict__ cd,
    const float* __restrict__ deg, float* __restrict__ posbuf,
    const _Float16* __restrict__ nw1f, const _Float16* __restrict__ nb1,
    const _Float16* __restrict__ nw2f, const _Float16* __restrict__ nb2,
    int n_tiles) {
  __shared__ NodeSmem sm[4];
  const int wave = threadIdx.x >> 5, lane = threadIdx.x & 31;
  const int nl = lane & 15, hi = lane >> 4;
  NodeSmem& S = sm[wave];
  int tile = blockIdx.x * 4 + wave;
  const bool active = tile < n_tiles;
  if (!active) tile = n_tiles - 1;
  const int row0 = tile * 16;
  const int col = nl * 8;

  #pragma unroll 4
  for (int e = 0; e < 16; ++e) {
    int row = row0 + e;
    if (hi == 0) {
      async_cp16(h16 + (size_t)row * HIDF + col, &S.u.ain[e][col]);
    } else {
      const float* ap = agg + (size_t)row * HIDF + col;
      union { _Float16 hh[8]; uint4 v; } t;
      #pragma unroll
      for (int j = 0; j < 8; ++j) t.hh[j] = (_Float16)ap[j];
      *(uint4*)&S.u.ain[e][128 + col] = t.v;
    }
  }
  wait_async();
  __syncthreads();

  const v8f vz = {0.f,0.f,0.f,0.f,0.f,0.f,0.f,0.f};
  v8f acc[8];
  #pragma unroll
  for (int nt = 0; nt < 8; ++nt) acc[nt] = vz;
  #pragma unroll
  for (int kt = 0; kt < 8; ++kt) {
    v16h a = lda(&S.u.ain[0][0], 264, kt * 32, lane);
    #pragma unroll
    for (int nt = 0; nt < 8; ++nt)
      acc[nt] = wmma32(a, ldb(nw1f, kt * 8 + nt, lane), acc[nt]);
  }
  #pragma unroll
  for (int nt = 0; nt < 8; ++nt) {
    int n = nt * 16 + nl;
    float bb = (float)nb1[n];
    #pragma unroll
    for (int r = 0; r < 8; ++r)
      S.u.t1[r + hi * 8][n] = (_Float16)silu_f(acc[nt][r] + bb);
  }
  __syncthreads();

  #pragma unroll
  for (int nt = 0; nt < 8; ++nt) acc[nt] = vz;
  #pragma unroll
  for (int kt = 0; kt < 4; ++kt) {
    v16h a = lda(&S.u.t1[0][0], 136, kt * 32, lane);
    #pragma unroll
    for (int nt = 0; nt < 8; ++nt)
      acc[nt] = wmma32(a, ldb(nw2f, kt * 8 + nt, lane), acc[nt]);
  }
  if (active) {
    #pragma unroll
    for (int nt = 0; nt < 8; ++nt) {
      int n = nt * 16 + nl;
      float bb = (float)nb2[n];
      #pragma unroll
      for (int r = 0; r < 8; ++r) {
        int row = row0 + r + hi * 8;
        size_t idx = (size_t)row * HIDF + n;
        float hv = h[idx] + acc[nt][r] + bb;   // residual
        h[idx] = hv;
        h16[idx] = (_Float16)hv;
      }
    }
    if (lane < 16) {
      int row = row0 + lane;
      float dg = fmaxf(deg[row], 1.0f);
      #pragma unroll
      for (int k2 = 0; k2 < 3; ++k2)
        posbuf[row * 3 + k2] += cd[row * 3 + k2] / dg;
    }
  }
}

// ---------------- output head ------------------------------------------------
__global__ void __launch_bounds__(128) out_kernel(
    const _Float16* __restrict__ h16,
    const _Float16* __restrict__ w1f, const _Float16* __restrict__ b1,
    const _Float16* __restrict__ w2f, const _Float16* __restrict__ b2,
    float* __restrict__ out, int n_tiles) {
  __shared__ __align__(16) _Float16 sA[4][16][136];
  const int wave = threadIdx.x >> 5, lane = threadIdx.x & 31;
  const int nl = lane & 15, hi = lane >> 4;
  int tile = blockIdx.x * 4 + wave;
  const bool active = tile < n_tiles;
  if (!active) tile = n_tiles - 1;
  const int row0 = tile * 16;

  #pragma unroll
  for (int e = 0; e < 8; ++e) {
    int row = row0 + e * 2 + hi;
    async_cp16(h16 + (size_t)row * HIDF + nl * 8, &sA[wave][e * 2 + hi][nl * 8]);
  }
  wait_async();
  __syncthreads();

  const v8f vz = {0.f,0.f,0.f,0.f,0.f,0.f,0.f,0.f};
  v8f acc[8];
  #pragma unroll
  for (int nt = 0; nt < 8; ++nt) acc[nt] = vz;
  #pragma unroll
  for (int kt = 0; kt < 4; ++kt) {
    v16h a = lda(&sA[wave][0][0], 136, kt * 32, lane);
    #pragma unroll
    for (int nt = 0; nt < 8; ++nt)
      acc[nt] = wmma32(a, ldb(w1f, kt * 8 + nt, lane), acc[nt]);
  }
  __syncthreads();
  #pragma unroll
  for (int nt = 0; nt < 8; ++nt) {
    int n = nt * 16 + nl;
    float bb = (float)b1[n];
    #pragma unroll
    for (int r = 0; r < 8; ++r)
      sA[wave][r + hi * 8][n] = (_Float16)silu_f(acc[nt][r] + bb);
  }
  __syncthreads();

  v8f acc2[4];
  #pragma unroll
  for (int nt = 0; nt < 4; ++nt) acc2[nt] = vz;
  #pragma unroll
  for (int kt = 0; kt < 4; ++kt) {
    v16h a = lda(&sA[wave][0][0], 136, kt * 32, lane);
    #pragma unroll
    for (int nt = 0; nt < 4; ++nt)
      acc2[nt] = wmma32(a, ldb(w2f, kt * 4 + nt, lane), acc2[nt]);
  }
  if (active) {
    #pragma unroll
    for (int nt = 0; nt < 4; ++nt) {
      int n = nt * 16 + nl;
      float bb = (float)b2[n];
      #pragma unroll
      for (int r = 0; r < 8; ++r) {
        int row = row0 + r + hi * 8;
        out[(size_t)row * 64 + n] = acc2[nt][r] + bb;
      }
    }
  }
}

// ---------------- host side --------------------------------------------------
extern "C" void kernel_launch(void* const* d_in, const int* in_sizes, int n_in,
                              void* d_out, int out_size, void* d_ws, size_t ws_size,
                              hipStream_t stream) {
  const int N = NN, E = NE;
  const float *x = nullptr, *pos = nullptr;
  const int* edge = nullptr;
  const float *embed_w = nullptr, *embed_b = nullptr;
  const float *out_w1 = nullptr, *out_b1 = nullptr, *out_w2 = nullptr, *out_b2 = nullptr;
  const float *e_w1[4], *e_b1[4], *e_w2[4], *e_b2[4];
  const float *n_w1[4], *n_b1[4], *n_w2[4], *n_b2[4];
  const float *c_w1[4], *c_b1[4], *c_w2[4];
  auto F = [&](int i) { return (const float*)d_in[i]; };

  if (n_in >= 2 && in_sizes[1] == 3 * N) {
    // insertion (dict) order: x, pos, edge_index, embed_w/b, out_w1/b1/w2/b2, layers...
    x = F(0); pos = F(1); edge = (const int*)d_in[2];
    embed_w = F(3); embed_b = F(4);
    out_w1 = F(5); out_b1 = F(6); out_w2 = F(7); out_b2 = F(8);
    for (int i = 0; i < 4; ++i) {
      int b = 9 + 11 * i;
      e_w1[i] = F(b + 0); e_b1[i] = F(b + 1); e_w2[i] = F(b + 2); e_b2[i] = F(b + 3);
      n_w1[i] = F(b + 4); n_b1[i] = F(b + 5); n_w2[i] = F(b + 6); n_b2[i] = F(b + 7);
      c_w1[i] = F(b + 8); c_b1[i] = F(b + 9); c_w2[i] = F(b + 10);
    }
  } else {
    // jax pytree (sorted-keys) order
    edge = (const int*)d_in[0];
    embed_b = F(1); embed_w = F(2);
    for (int i = 0; i < 4; ++i) {
      int b = 3 + 11 * i;
      c_b1[i] = F(b + 0); c_w1[i] = F(b + 1); c_w2[i] = F(b + 2);
      e_b1[i] = F(b + 3); e_b2[i] = F(b + 4); e_w1[i] = F(b + 5); e_w2[i] = F(b + 6);
      n_b1[i] = F(b + 7); n_b2[i] = F(b + 8); n_w1[i] = F(b + 9); n_w2[i] = F(b + 10);
    }
    out_b1 = F(47); out_b2 = F(48); out_w1 = F(49); out_w2 = F(50);
    pos = F(51); x = F(52);
  }

  // -------- workspace carve --------
  char* wsp = (char*)d_ws;
  size_t off = 0;
  auto alloc = [&](size_t bytes) -> void* {
    void* p = wsp + off;
    off += (bytes + 255) & ~(size_t)255;
    return p;
  };
  float*    h   = (float*)alloc((size_t)N * HIDF * sizeof(float));
  _Float16* h16 = (_Float16*)alloc((size_t)N * HIDF * sizeof(_Float16));
  float*    agg = (float*)alloc((size_t)N * HIDF * sizeof(float));
  float*    cdv = (float*)alloc((size_t)N * 3 * sizeof(float));
  float*    deg = (float*)alloc((size_t)N * sizeof(float));
  _Float16* embed_wf = (_Float16*)alloc(64 * 128 * 2);
  _Float16* embed_bf = (_Float16*)alloc(128 * 2);
  _Float16* ow1f = (_Float16*)alloc(128 * 128 * 2);
  _Float16* ob1f = (_Float16*)alloc(128 * 2);
  _Float16* ow2f = (_Float16*)alloc(128 * 64 * 2);
  _Float16* ob2f = (_Float16*)alloc(64 * 2);
  _Float16 *ew1f[4], *ew1lastf[4], *eb1f[4], *ew2f[4], *eb2f[4];
  _Float16 *cw1f[4], *cb1f[4], *cw2f[4], *nw1f[4], *nb1f[4], *nw2f[4], *nb2f[4];
  for (int i = 0; i < 4; ++i) {
    ew1f[i] = (_Float16*)alloc(256 * 128 * 2);
    ew1lastf[i] = (_Float16*)alloc(128 * 2);
    eb1f[i] = (_Float16*)alloc(128 * 2);
    ew2f[i] = (_Float16*)alloc(128 * 128 * 2);
    eb2f[i] = (_Float16*)alloc(128 * 2);
    cw1f[i] = (_Float16*)alloc(128 * 128 * 2);
    cb1f[i] = (_Float16*)alloc(128 * 2);
    cw2f[i] = (_Float16*)alloc(128 * 2);
    nw1f[i] = (_Float16*)alloc(256 * 128 * 2);
    nb1f[i] = (_Float16*)alloc(128 * 2);
    nw2f[i] = (_Float16*)alloc(128 * 128 * 2);
    nb2f[i] = (_Float16*)alloc(128 * 2);
  }
  (void)ws_size; (void)out_size;

  auto repack = [&](const float* W, _Float16* Wf, int K, int Ncols) {
    int total = (K / 32) * (Ncols / 16) * 512;
    repack_w<<<(total + 255) / 256, 256, 0, stream>>>(W, Wf, K / 32, Ncols / 16);
  };
  auto cvt = [&](const float* s, _Float16* d, int n) {
    cvt_f16<<<(n + 255) / 256, 256, 0, stream>>>(s, d, n);
  };

  repack(embed_w, embed_wf, 64, 128);  cvt(embed_b, embed_bf, 128);
  repack(out_w1, ow1f, 128, 128);      cvt(out_b1, ob1f, 128);
  repack(out_w2, ow2f, 128, 64);       cvt(out_b2, ob2f, 64);
  for (int i = 0; i < 4; ++i) {
    repack(e_w1[i], ew1f[i], 256, 128);
    cvt(e_w1[i] + 256 * 128, ew1lastf[i], 128);   // d2 row of e_w1
    cvt(e_b1[i], eb1f[i], 128);
    repack(e_w2[i], ew2f[i], 128, 128); cvt(e_b2[i], eb2f[i], 128);
    repack(c_w1[i], cw1f[i], 128, 128); cvt(c_b1[i], cb1f[i], 128);
    cvt(c_w2[i], cw2f[i], 128);
    repack(n_w1[i], nw1f[i], 256, 128); cvt(n_b1[i], nb1f[i], 128);
    repack(n_w2[i], nw2f[i], 128, 128); cvt(n_b2[i], nb2f[i], 128);
  }

  const int* src = edge;
  const int* dst = edge + E;
  float* posbuf = (float*)d_out + (size_t)N * 64;   // pos output region, used live

  (void)hipMemsetAsync(deg, 0, (size_t)N * sizeof(float), stream);
  deg_kernel<<<(E + 255) / 256, 256, 0, stream>>>(dst, deg, E);
  (void)hipMemcpyAsync(posbuf, pos, (size_t)N * 3 * sizeof(float),
                       hipMemcpyDeviceToDevice, stream);

  const int n_tiles = N / 16;              // 625
  const int nblocks = (n_tiles + 3) / 4;   // 4 waves/block, 16 rows/wave
  embed_kernel<<<nblocks, 128, 0, stream>>>(x, embed_wf, embed_bf, h, h16, n_tiles);

  const int eblocks = E / 64;              // 16 edges/wave * 4 waves
  for (int i = 0; i < 4; ++i) {
    (void)hipMemsetAsync(agg, 0, (size_t)N * HIDF * sizeof(float), stream);
    (void)hipMemsetAsync(cdv, 0, (size_t)N * 3 * sizeof(float), stream);
    edge_kernel<<<eblocks, 128, 0, stream>>>(src, dst, posbuf, h16, agg, cdv,
        ew1f[i], ew1lastf[i], eb1f[i], ew2f[i], eb2f[i], cw1f[i], cb1f[i], cw2f[i]);
    node_kernel<<<nblocks, 128, 0, stream>>>(h, h16, agg, cdv, deg, posbuf,
        nw1f[i], nb1f[i], nw2f[i], nb2f[i], n_tiles);
  }
  out_kernel<<<nblocks, 128, 0, stream>>>(h16, ow1f, ob1f, ow2f, ob2f,
                                          (float*)d_out, n_tiles);
}
